// Informer_2405181686010
// MI455X (gfx1250) — compile-verified, matrix operand order checked by reading, not run on hardware
//
#include <hip/hip_runtime.h>
#include <hip/hip_bf16.h>
#include <stdint.h>

// ---------------------------------------------------------------------------
// Informer forward on gfx1250 (MI455X).
// All dense GEMMs run on v_wmma_f32_16x16x32_f16 (f16 in, f32 accumulate),
// double-buffered LDS pipeline, one workgroup barrier per K-panel.
// fp32 -> f16 conversion happens during LDS staging (no extra passes).
// ProbSparse attention replicates JAX threefry2x32 exactly for the sampling
// indices (span is a power of two, so randint == bits & (LK-1)).
// ---------------------------------------------------------------------------

typedef _Float16 f16;
typedef __attribute__((ext_vector_type(16))) _Float16 v16h;
typedef __attribute__((ext_vector_type(8)))  _Float16 v8h;
typedef __attribute__((ext_vector_type(2)))  _Float16 v2h;
typedef __attribute__((ext_vector_type(8)))  float    v8f;

#define NEG_INF (-__builtin_inff())

#define BM 128
#define BN 128
#define BK 32
#define LDSS 40   // f16 row stride: 80 bytes, keeps 16B-aligned rows

#define FLAG_BIAS  1
#define FLAG_RESID 2
#define FLAG_GELU  4

// ---------------------------------------------------------------------------
// WMMA GEMM: C[M,N] = A[M,K] * B[K,N] (+bias) (+residual) (optional GELU)
// M % 128 == 0, N % 128 == 0, K % 32 == 0 (guaranteed by the model dims).
// Block: 256 threads = 8 waves arranged 4(M) x 2(N); each wave: 2x4 16x16
// accumulators = 32x64 sub-tile. FLAGS is compile-time -> branch-free epilogue.
// ---------------------------------------------------------------------------
template <int FLAGS>
__global__ void __launch_bounds__(256)
wmma_gemm_kernel(const float* __restrict__ A, const float* __restrict__ Bm,
                 const float* __restrict__ bias, const float* __restrict__ resid,
                 float* __restrict__ C, int M, int N, int K)
{
    __shared__ __align__(16) f16 As[2][BM][LDSS];
    __shared__ __align__(16) f16 Bs[2][BN][LDSS];   // [col][k] (transposed)

    const int t    = threadIdx.x;
    const int wave = t >> 5;
    const int lane = t & 31;
    const int wm   = wave & 3;        // 0..3 -> rows 32*wm
    const int wn   = wave >> 2;       // 0..1 -> cols 64*wn
    const int row0 = blockIdx.y * BM;
    const int col0 = blockIdx.x * BN;

    v8f acc[2][4];
#pragma unroll
    for (int i = 0; i < 2; ++i)
#pragma unroll
        for (int j = 0; j < 4; ++j)
#pragma unroll
            for (int e = 0; e < 8; ++e) acc[i][j][e] = 0.0f;

    // A staging: thread -> (row, 16-col half-panel); contiguous b128 global.
    const int la_r = t >> 1;          // 0..127
    const int la_c = (t & 1) * 16;    // 0 or 16
    // B staging: thread -> (k-row pair, 8 cols); pairs pack into ds_store_b32.
    const int lb_r = (t >> 4) * 2;    // 0,2,..,30
    const int lb_c = (t & 15) * 8;    // 0..120

    const int rsel = lane & 15;
    const int koff = (lane >> 4) * 8; // 0 or 8 (ISA 16-bit A/B fragment layout)

    float ar[16];                     // staged A row chunk
    float br0[8], br1[8];             // staged B row pair

    auto gload = [&](int k0) {
        const float4* ap = (const float4*)(A + (size_t)(row0 + la_r) * K + (k0 + la_c));
#pragma unroll
        for (int j = 0; j < 4; ++j) {
            float4 v = ap[j];
            ar[4 * j + 0] = v.x; ar[4 * j + 1] = v.y;
            ar[4 * j + 2] = v.z; ar[4 * j + 3] = v.w;
        }
        const float4* bp0 = (const float4*)(Bm + (size_t)(k0 + lb_r) * N + (col0 + lb_c));
        const float4* bp1 = (const float4*)(Bm + (size_t)(k0 + lb_r + 1) * N + (col0 + lb_c));
#pragma unroll
        for (int j = 0; j < 2; ++j) {
            float4 v0 = bp0[j], v1 = bp1[j];
            br0[4 * j + 0] = v0.x; br0[4 * j + 1] = v0.y; br0[4 * j + 2] = v0.z; br0[4 * j + 3] = v0.w;
            br1[4 * j + 0] = v1.x; br1[4 * j + 1] = v1.y; br1[4 * j + 2] = v1.z; br1[4 * j + 3] = v1.w;
        }
    };

    auto sstore = [&](int buf) {
        v8h h0, h1;
#pragma unroll
        for (int j = 0; j < 8; ++j) { h0[j] = (f16)ar[j]; h1[j] = (f16)ar[8 + j]; }
        *(v8h*)&As[buf][la_r][la_c]     = h0;
        *(v8h*)&As[buf][la_r][la_c + 8] = h1;
#pragma unroll
        for (int j = 0; j < 8; ++j) {
            v2h p; p[0] = (f16)br0[j]; p[1] = (f16)br1[j];
            *(v2h*)&Bs[buf][lb_c + j][lb_r] = p;   // consecutive k: one b32 store
        }
    };

    const int nk = K / BK;
    gload(0);
    sstore(0);
    __syncthreads();

    for (int tk = 0; tk < nk; ++tk) {
        const bool has_next = (tk + 1 < nk);
        if (has_next) gload((tk + 1) * BK);    // overlap VMEM with WMMA

        const int cur = tk & 1;
        v16h afr[2], bfr[4];
#pragma unroll
        for (int fm = 0; fm < 2; ++fm) {
            const f16* p = &As[cur][wm * 32 + fm * 16 + rsel][0];
            union { v16h v; v8h h[2]; } u;
            u.h[0] = *(const v8h*)(p + koff);
            u.h[1] = *(const v8h*)(p + 16 + koff);
            afr[fm] = u.v;
        }
#pragma unroll
        for (int fn = 0; fn < 4; ++fn) {
            const f16* p = &Bs[cur][wn * 64 + fn * 16 + rsel][0];
            union { v16h v; v8h h[2]; } u;
            u.h[0] = *(const v8h*)(p + koff);
            u.h[1] = *(const v8h*)(p + 16 + koff);
            bfr[fn] = u.v;
        }
#pragma unroll
        for (int fm = 0; fm < 2; ++fm)
#pragma unroll
            for (int fn = 0; fn < 4; ++fn)
                acc[fm][fn] = __builtin_amdgcn_wmma_f32_16x16x32_f16(
                    false, afr[fm], false, bfr[fn],
                    (short)0, acc[fm][fn], false, false);

        if (has_next) sstore((tk + 1) & 1);    // other buffer: no reader conflict
        __syncthreads();                       // single barrier per K-panel
    }

    // Epilogue: C layout per ISA (lanes 0-15: M=v, lanes 16-31: M=8+v)
    const int roff = (lane >> 4) * 8;
#pragma unroll
    for (int fm = 0; fm < 2; ++fm) {
#pragma unroll
        for (int fn = 0; fn < 4; ++fn) {
            const int colg = col0 + wn * 64 + fn * 16 + rsel;
            const float bv = (FLAGS & FLAG_BIAS) ? bias[colg] : 0.0f;
#pragma unroll
            for (int v = 0; v < 8; ++v) {
                const int rowg = row0 + wm * 32 + fm * 16 + roff + v;
                float x = acc[fm][fn][v] + bv;
                if (FLAGS & FLAG_RESID) x += resid[(size_t)rowg * N + colg];
                if (FLAGS & FLAG_GELU)  x = 0.5f * x * (1.0f + erff(x * 0.70710678118654752f));
                C[(size_t)rowg * N + colg] = x;
            }
        }
    }
}

// ---------------------------------------------------------------------------
// Threefry2x32 (JAX-exact) for ProbSparse sampling indices.
// ---------------------------------------------------------------------------
__device__ __forceinline__ void tf_round4(uint32_t& x0, uint32_t& x1,
                                          const int r0, const int r1, const int r2, const int r3)
{
    const int rr[4] = {r0, r1, r2, r3};
#pragma unroll
    for (int i = 0; i < 4; ++i) {
        x0 += x1;
        x1 = (x1 << rr[i]) | (x1 >> (32 - rr[i]));
        x1 ^= x0;
    }
}

__device__ __forceinline__ void threefry2x32(uint32_t k0, uint32_t k1,
                                             uint32_t x0, uint32_t x1,
                                             uint32_t& o0, uint32_t& o1)
{
    uint32_t k2 = k0 ^ k1 ^ 0x1BD11BDAu;
    x0 += k0; x1 += k1;
    tf_round4(x0, x1, 13, 15, 26, 6);  x0 += k1; x1 += k2 + 1u;
    tf_round4(x0, x1, 17, 29, 16, 24); x0 += k2; x1 += k0 + 2u;
    tf_round4(x0, x1, 13, 15, 26, 6);  x0 += k0; x1 += k1 + 3u;
    tf_round4(x0, x1, 17, 29, 16, 24); x0 += k1; x1 += k2 + 4u;
    tf_round4(x0, x1, 13, 15, 26, 6);  x0 += k2; x1 += k0 + 5u;
    o0 = x0; o1 = x1;
}

// idx[i] = randint(fold_in(key(42), cid), (n,), 0, LK) with LK power-of-two.
// JAX counter mode: counts split into halves; bits[j]=out0(j, half+j), bits[half+j]=out1.
__global__ void sample_idx_kernel(int* __restrict__ idx, int n, int mask, int cid)
{
    int j = blockIdx.x * blockDim.x + threadIdx.x;
    int half = n >> 1;
    if (j >= half) return;
    uint32_t fk0, fk1;
    threefry2x32(0u, 42u, 0u, (uint32_t)cid, fk0, fk1);   // fold_in
    uint32_t b0, b1;
    threefry2x32(fk0, fk1, (uint32_t)j, (uint32_t)(half + j), b0, b1);
    idx[j]        = (int)(b0 & (uint32_t)mask);
    idx[half + j] = (int)(b1 & (uint32_t)mask);
}

// ---------------------------------------------------------------------------
// Data embedding: circular token conv (K=3, Cin->512) + time features + PE.
// ---------------------------------------------------------------------------
__global__ void embed_kernel(const float* __restrict__ x, const float* __restrict__ xmark,
                             const float* __restrict__ token_w, const float* __restrict__ timef_w,
                             float* __restrict__ out, int Bn, int L, int Cin)
{
    int i = blockIdx.x * blockDim.x + threadIdx.x;
    int total = Bn * L * 512;
    if (i >= total) return;
    int d  = i & 511;
    int bl = i >> 9;
    int l  = bl % L;
    int b  = bl / L;

    float acc = 0.0f;
#pragma unroll
    for (int tt = 0; tt < 3; ++tt) {
        int ls = l + tt - 1;
        ls = (ls + L) % L;                       // circular pad
        const float* xp = x + (size_t)(b * L + ls) * Cin;
        const float* wp = token_w + (size_t)tt * Cin * 512 + d;
        for (int c = 0; c < Cin; ++c) acc += xp[c] * wp[(size_t)c * 512];
    }
    const float* mp = xmark + (size_t)(b * L + l) * 4;
#pragma unroll
    for (int m = 0; m < 4; ++m) acc += mp[m] * timef_w[m * 512 + d];

    int   pi   = d >> 1;
    float freq = expf(-(float)(2 * pi) * logf(10000.0f) / 512.0f);
    float ang  = (float)l * freq;
    acc += (d & 1) ? cosf(ang) : sinf(ang);
    out[(size_t)bl * 512 + d] = acc;
}

// ---------------------------------------------------------------------------
// ProbSparse metric: M[b,h,l] = max_s(q.k_s) - sum_s(q.k_s)/LK
// q,k layout: [B, L, H*64] row-major (natural GEMM output).
// ---------------------------------------------------------------------------
__global__ void metric_kernel(const float* __restrict__ q, const float* __restrict__ k,
                              const int* __restrict__ idx, float* __restrict__ Mout,
                              int Bn, int LQ, int LK, int U)
{
    int i = blockIdx.x * blockDim.x + threadIdx.x;
    int total = Bn * 8 * LQ;
    if (i >= total) return;
    int l  = i % LQ;
    int bh = i / LQ;
    int h  = bh & 7;
    int b  = bh >> 3;
    const float* qp = q + (size_t)(b * LQ + l) * 512 + h * 64;
    float mx = NEG_INF, sm = 0.0f;
    for (int s = 0; s < U; ++s) {
        int ki = idx[l * U + s];
        const float* kp = k + (size_t)(b * LK + ki) * 512 + h * 64;
        float d = 0.0f;
#pragma unroll
        for (int e = 0; e < 64; ++e) d += qp[e] * kp[e];
        mx = fmaxf(mx, d);
        sm += d;
    }
    Mout[(size_t)bh * LQ + l] = mx - sm / (float)LK;
}

// ---------------------------------------------------------------------------
// Exact top-k (u selections, tie -> lowest index, like jax.lax.top_k).
// One block per (b,h).
// ---------------------------------------------------------------------------
__global__ void topk_kernel(const float* __restrict__ Mi, int* __restrict__ topIdx,
                            int LQ, int U)
{
    __shared__ float sm[1024];
    __shared__ float bv[256];
    __shared__ int   bix[256];
    int bh = blockIdx.x;
    int t  = threadIdx.x;
    for (int i = t; i < LQ; i += 256) sm[i] = Mi[(size_t)bh * LQ + i];
    __syncthreads();
    for (int sel = 0; sel < U; ++sel) {
        float best = NEG_INF; int bidx = 0x7fffffff;
        for (int i = t; i < LQ; i += 256) {
            float v = sm[i];
            if (v > best || (v == best && i < bidx)) { best = v; bidx = i; }
        }
        bv[t] = best; bix[t] = bidx;
        __syncthreads();
        for (int s = 128; s > 0; s >>= 1) {
            if (t < s) {
                if (bv[t + s] > bv[t] || (bv[t + s] == bv[t] && bix[t + s] < bix[t])) {
                    bv[t] = bv[t + s]; bix[t] = bix[t + s];
                }
            }
            __syncthreads();
        }
        if (t == 0) {
            topIdx[(size_t)bh * U + sel] = bix[0];
            sm[bix[0]] = NEG_INF;
        }
        __syncthreads();
    }
}

// Context base (non-masked): broadcast mean of V over keys.
__global__ void ctx_mean_kernel(const float* __restrict__ v, float* __restrict__ ctx,
                                int LQ, int LK)
{
    int bh = blockIdx.x;
    int h  = bh & 7;
    int b  = bh >> 3;
    int d  = threadIdx.x;     // 64 threads
    float s = 0.0f;
    for (int l = 0; l < LK; ++l) s += v[(size_t)(b * LK + l) * 512 + h * 64 + d];
    s /= (float)LK;
    for (int l = 0; l < LQ; ++l) ctx[(size_t)(b * LQ + l) * 512 + h * 64 + d] = s;
}

// Context base (masked): cumulative sum of V along the sequence.
__global__ void ctx_cumsum_kernel(const float* __restrict__ v, float* __restrict__ ctx,
                                  int Bn, int L)
{
    int i = blockIdx.x * blockDim.x + threadIdx.x;
    if (i >= Bn * 8 * 64) return;
    int d  = i & 63;
    int bh = i >> 6;
    int h  = bh & 7;
    int b  = bh >> 3;
    float s = 0.0f;
    for (int l = 0; l < L; ++l) {
        s += v[(size_t)(b * L + l) * 512 + h * 64 + d];
        ctx[(size_t)(b * L + l) * 512 + h * 64 + d] = s;
    }
}

// Full attention for the top-u queries, scattered into ctx.
// One block (256 thr) per (b,h,u).
__global__ void topq_kernel(const float* __restrict__ q, const float* __restrict__ k,
                            const float* __restrict__ v, const int* __restrict__ topIdx,
                            float* __restrict__ ctx, int LQ, int LK, int U, int masked)
{
    __shared__ float sc[1024];
    __shared__ float red[256];
    int blk = blockIdx.x;
    int u   = blk % U;
    int bh  = blk / U;
    int h   = bh & 7;
    int b   = bh >> 3;
    int t   = threadIdx.x;
    int iu  = topIdx[(size_t)bh * U + u];
    const float* qp = q + (size_t)(b * LQ + iu) * 512 + h * 64;

    float lmax = NEG_INF;
    for (int kk = t; kk < LK; kk += 256) {
        const float* kp = k + (size_t)(b * LK + kk) * 512 + h * 64;
        float d = 0.0f;
#pragma unroll
        for (int e = 0; e < 64; ++e) d += qp[e] * kp[e];
        d *= 0.125f;                               // 1/sqrt(64)
        if (masked && kk > iu) d = NEG_INF;        // strict upper-tri mask
        sc[kk] = d;
        lmax = fmaxf(lmax, d);
    }
    red[t] = lmax; __syncthreads();
    for (int s = 128; s > 0; s >>= 1) { if (t < s) red[t] = fmaxf(red[t], red[t + s]); __syncthreads(); }
    float gmax = red[0]; __syncthreads();

    float lsum = 0.0f;
    for (int kk = t; kk < LK; kk += 256) {
        float e = expf(sc[kk] - gmax);
        sc[kk] = e;
        lsum += e;
    }
    red[t] = lsum; __syncthreads();
    for (int s = 128; s > 0; s >>= 1) { if (t < s) red[t] += red[t + s]; __syncthreads(); }
    float inv = 1.0f / red[0];
    __syncthreads();

    if (t < 64) {
        float o = 0.0f;
        for (int kk = 0; kk < LK; ++kk)
            o += sc[kk] * v[(size_t)(b * LK + kk) * 512 + h * 64 + t];
        ctx[(size_t)(b * LQ + iu) * 512 + h * 64 + t] = o * inv;
    }
}

// LayerNorm over last dim (512). One block per token, 256 threads x 2 elems.
__global__ void ln_kernel(const float* __restrict__ x, const float* __restrict__ g,
                          const float* __restrict__ b, float* __restrict__ out)
{
    __shared__ float red[256];
    int tok = blockIdx.x;
    int t   = threadIdx.x;
    const float* xp = x + (size_t)tok * 512;
    float a0 = xp[t], a1 = xp[t + 256];
    red[t] = a0 + a1; __syncthreads();
    for (int s = 128; s > 0; s >>= 1) { if (t < s) red[t] += red[t + s]; __syncthreads(); }
    float mu = red[0] * (1.0f / 512.0f);
    __syncthreads();
    float d0 = a0 - mu, d1 = a1 - mu;
    red[t] = d0 * d0 + d1 * d1; __syncthreads();
    for (int s = 128; s > 0; s >>= 1) { if (t < s) red[t] += red[t + s]; __syncthreads(); }
    float inv = rsqrtf(red[0] * (1.0f / 512.0f) + 1e-5f);
    out[(size_t)tok * 512 + t]       = d0 * inv * g[t]       + b[t];
    out[(size_t)tok * 512 + t + 256] = d1 * inv * g[t + 256] + b[t + 256];
}

// im2col for the distil conv (circular, K=3): col[token][t*512+c] = x[wrap(l+t-1)][c]
__global__ void im2col_kernel(const float* __restrict__ x, float* __restrict__ col,
                              int Bn, int L)
{
    int i = blockIdx.x * blockDim.x + threadIdx.x;
    int total = Bn * L * 1536;
    if (i >= total) return;
    int tc = i % 1536;
    int bl = i / 1536;
    int l  = bl % L;
    int b  = bl / L;
    int tt = tc / 512;
    int c  = tc % 512;
    int ls = l + tt - 1;
    ls = (ls + L) % L;
    col[(size_t)bl * 1536 + tc] = x[(size_t)(b * L + ls) * 512 + c];
}

// BatchNorm + ELU + maxpool(k=3, stride 2, -inf pad)
__global__ void bn_elu_pool_kernel(const float* __restrict__ y, const float* __restrict__ g,
                                   const float* __restrict__ bb, const float* __restrict__ rm,
                                   const float* __restrict__ rv, float* __restrict__ out,
                                   int Bn, int Lin)
{
    int i = blockIdx.x * blockDim.x + threadIdx.x;
    int Lo = Lin >> 1;
    int total = Bn * Lo * 512;
    if (i >= total) return;
    int c  = i & 511;
    int bo = i >> 9;
    int oi = bo % Lo;
    int b  = bo / Lo;
    float inv = rsqrtf(rv[c] + 1e-5f);
    float best = NEG_INF;
#pragma unroll
    for (int dj = 0; dj < 3; ++dj) {
        int jj = 2 * oi - 1 + dj;
        if (jj >= 0 && jj < Lin) {
            float tv = (y[(size_t)(b * Lin + jj) * 512 + c] - rm[c]) * inv * g[c] + bb[c];
            tv = tv > 0.0f ? tv : (expf(tv) - 1.0f);
            best = fmaxf(best, tv);
        }
    }
    out[(size_t)(b * Lo + oi) * 512 + c] = best;
}

// Final projection 512 -> 7, emitting only the last PRED_LEN rows.
__global__ void proj_kernel(const float* __restrict__ x, const float* __restrict__ w,
                            const float* __restrict__ bias, float* __restrict__ out,
                            int Bn, int L, int P)
{
    int i = blockIdx.x * blockDim.x + threadIdx.x;
    int total = Bn * P * 7;
    if (i >= total) return;
    int c  = i % 7;
    int bt = i / 7;
    int tp = bt % P;
    int b  = bt / P;
    int row = L - P + tp;
    const float* xp = x + (size_t)(b * L + row) * 512;
    float s = bias[c];
    for (int kk = 0; kk < 512; ++kk) s += xp[kk] * w[kk * 7 + c];
    out[(size_t)(b * P + tp) * 7 + c] = s;
}

// ---------------------------------------------------------------------------
// Host orchestration
// ---------------------------------------------------------------------------
static void gemm(const float* A, const float* B, const float* bias, const float* resid,
                 float* C, int M, int N, int K, int flags, hipStream_t s)
{
    dim3 g(N / 128, M / 128);
    switch (flags) {
        case FLAG_BIAS:
            wmma_gemm_kernel<FLAG_BIAS><<<g, 256, 0, s>>>(A, B, bias, resid, C, M, N, K);
            break;
        case FLAG_BIAS | FLAG_RESID:
            wmma_gemm_kernel<FLAG_BIAS | FLAG_RESID><<<g, 256, 0, s>>>(A, B, bias, resid, C, M, N, K);
            break;
        case FLAG_BIAS | FLAG_GELU:
            wmma_gemm_kernel<FLAG_BIAS | FLAG_GELU><<<g, 256, 0, s>>>(A, B, bias, resid, C, M, N, K);
            break;
        default:
            wmma_gemm_kernel<0><<<g, 256, 0, s>>>(A, B, bias, resid, C, M, N, K);
            break;
    }
}

struct AttnW { const float *wq, *bq, *wk, *bk, *wv, *bv, *wo, *bo; };

static void run_attention(const AttnW& w, const float* xq, const float* xkv,
                          int Bn, int LQ, int LK, bool masked, int cid,
                          const float* resid, float* out,
                          float* Qb, float* Kb, float* Vb, float* CTX,
                          float* Mb, int* idxb, int* topb, hipStream_t s)
{
    const int U = 35;  // FACTOR * ceil(ln(L)) for L in {512,1024}
    gemm(xq,  w.wq, w.bq, nullptr, Qb, Bn * LQ, 512, 512, FLAG_BIAS, s);
    gemm(xkv, w.wk, w.bk, nullptr, Kb, Bn * LK, 512, 512, FLAG_BIAS, s);
    gemm(xkv, w.wv, w.bv, nullptr, Vb, Bn * LK, 512, 512, FLAG_BIAS, s);
    int n = LQ * U, half = n / 2;
    sample_idx_kernel<<<(half + 255) / 256, 256, 0, s>>>(idxb, n, LK - 1, cid);
    int tot = Bn * 8 * LQ;
    metric_kernel<<<(tot + 255) / 256, 256, 0, s>>>(Qb, Kb, idxb, Mb, Bn, LQ, LK, U);
    topk_kernel<<<Bn * 8, 256, 0, s>>>(Mb, topb, LQ, U);
    if (masked) {
        int t2 = Bn * 8 * 64;
        ctx_cumsum_kernel<<<(t2 + 255) / 256, 256, 0, s>>>(Vb, CTX, Bn, LK);
    } else {
        ctx_mean_kernel<<<Bn * 8, 64, 0, s>>>(Vb, CTX, LQ, LK);
    }
    topq_kernel<<<Bn * 8 * U, 256, 0, s>>>(Qb, Kb, Vb, topb, CTX, LQ, LK, U, masked ? 1 : 0);
    gemm(CTX, w.wo, w.bo, resid, out, Bn * LQ, 512, 512, FLAG_BIAS | FLAG_RESID, s);
}

static void run_ffn(const float* x, const float* w1, const float* b1,
                    const float* w2, const float* b2, float* mid, float* out,
                    int M, hipStream_t s)
{
    gemm(x, w1, b1, nullptr, mid, M, 2048, 512, FLAG_BIAS | FLAG_GELU, s);
    gemm(mid, w2, b2, x, out, M, 512, 2048, FLAG_BIAS | FLAG_RESID, s);
}

extern "C" void kernel_launch(void* const* d_in, const int* in_sizes, int n_in,
                              void* d_out, int out_size, void* d_ws, size_t ws_size,
                              hipStream_t stream)
{
    (void)in_sizes; (void)n_in; (void)out_size; (void)ws_size;

    const float* x_enc      = (const float*)d_in[0];
    const float* x_mark_enc = (const float*)d_in[1];
    const float* x_dec      = (const float*)d_in[2];
    const float* x_mark_dec = (const float*)d_in[3];
    // params in JAX pytree (sorted-dict) flatten order:
    //  4 dec_emb.timef_w  5 dec_emb.token_w
    //  6 decL.conv1_b 7 conv1_w 8 conv2_b 9 conv2_w
    // 10-17 decL.cross_attn {bk,bo,bq,bv,wk,wo,wq,wv}
    // 18 ln1_b 19 ln1_g 20 ln2_b 21 ln2_g 22 ln3_b 23 ln3_g
    // 24-31 decL.self_attn {bk,bo,bq,bv,wk,wo,wq,wv}
    // 32 dec_norm_b 33 dec_norm_g
    // 34 conv.b 35 bn_b 36 bn_g 37 bn_rm 38 bn_rv 39 conv.w
    // 40 enc_emb.timef_w 41 enc_emb.token_w
    // 42-49 encL0.attn {bk,bo,bq,bv,wk,wo,wq,wv}
    // 50 conv1_b 51 conv1_w 52 conv2_b 53 conv2_w 54 ln1_b 55 ln1_g 56 ln2_b 57 ln2_g
    // 58-65 encL1.attn, 66-73 encL1 ffn/ln (same pattern)
    // 74 enc_norm_b 75 enc_norm_g 76 proj_b 77 proj_w
    auto F = [&](int i) { return (const float*)d_in[i]; };

    char* ws = (char*)d_ws;
    const size_t MB = (size_t)1 << 20;
    float* ENCX = (float*)(ws + 0 * MB);    // 8 MB: 4096x512
    float* A2   = (float*)(ws + 8 * MB);    // 8 MB
    float* T1   = (float*)(ws + 16 * MB);   // 8 MB
    float* Qb   = (float*)(ws + 24 * MB);   // 8 MB
    float* Kb   = (float*)(ws + 32 * MB);   // 8 MB
    float* Vb   = (float*)(ws + 40 * MB);   // 8 MB
    float* CTX  = (float*)(ws + 48 * MB);   // 8 MB
    float* MID  = (float*)(ws + 56 * MB);   // 32 MB: FFN mid / im2col
    float* ENCF = (float*)(ws + 88 * MB);   // 4 MB: final encoder output
    float* Mb   = (float*)(ws + 92 * MB);            // 128 KB metric
    int*   IDXb = (int*)(ws + 92 * MB + 512 * 1024); // 144 KB sample idx
    int*   TOPb = (int*)(ws + 92 * MB + 768 * 1024); // 4.5 KB top-k idx

    // ---- Encoder ----
    int totE = 4 * 1024 * 512;
    embed_kernel<<<(totE + 255) / 256, 256, 0, stream>>>(x_enc, x_mark_enc, F(41), F(40), ENCX, 4, 1024, 7);

    AttnW e0 = {F(48), F(44), F(46), F(42), F(49), F(45), F(47), F(43)};
    run_attention(e0, ENCX, ENCX, 4, 1024, 1024, false, 0, ENCX, T1, Qb, Kb, Vb, CTX, Mb, IDXb, TOPb, stream);
    ln_kernel<<<4096, 256, 0, stream>>>(T1, F(55), F(54), A2);
    run_ffn(A2, F(51), F(50), F(53), F(52), MID, T1, 4096, stream);
    ln_kernel<<<4096, 256, 0, stream>>>(T1, F(57), F(56), ENCX);

    // distil: im2col + GEMM(K=1536) + BN/ELU/maxpool -> L: 1024 -> 512
    int totC = 4 * 1024 * 1536;
    im2col_kernel<<<(totC + 255) / 256, 256, 0, stream>>>(ENCX, MID, 4, 1024);
    gemm(MID, F(39), F(34), nullptr, T1, 4096, 512, 1536, FLAG_BIAS, stream);
    int totP = 4 * 512 * 512;
    bn_elu_pool_kernel<<<(totP + 255) / 256, 256, 0, stream>>>(T1, F(36), F(35), F(37), F(38), A2, 4, 1024);

    AttnW e1 = {F(64), F(60), F(62), F(58), F(65), F(61), F(63), F(59)};
    run_attention(e1, A2, A2, 4, 512, 512, false, 1, A2, T1, Qb, Kb, Vb, CTX, Mb, IDXb, TOPb, stream);
    ln_kernel<<<2048, 256, 0, stream>>>(T1, F(71), F(70), ENCX);
    run_ffn(ENCX, F(67), F(66), F(69), F(68), MID, T1, 2048, stream);
    ln_kernel<<<2048, 256, 0, stream>>>(T1, F(73), F(72), A2);
    ln_kernel<<<2048, 256, 0, stream>>>(A2, F(75), F(74), ENCF);   // encoder final norm

    // ---- Decoder ----
    int totD = 4 * 512 * 512;
    embed_kernel<<<(totD + 255) / 256, 256, 0, stream>>>(x_dec, x_mark_dec, F(5), F(4), ENCX, 4, 512, 7);

    AttnW ds = {F(30), F(26), F(28), F(24), F(31), F(27), F(29), F(25)};
    run_attention(ds, ENCX, ENCX, 4, 512, 512, true, 2, ENCX, T1, Qb, Kb, Vb, CTX, Mb, IDXb, TOPb, stream);
    ln_kernel<<<2048, 256, 0, stream>>>(T1, F(19), F(18), A2);

    AttnW dc = {F(16), F(12), F(14), F(10), F(17), F(13), F(15), F(11)};
    run_attention(dc, A2, ENCF, 4, 512, 512, false, 3, A2, T1, Qb, Kb, Vb, CTX, Mb, IDXb, TOPb, stream);
    ln_kernel<<<2048, 256, 0, stream>>>(T1, F(21), F(20), ENCX);
    run_ffn(ENCX, F(7), F(6), F(9), F(8), MID, T1, 2048, stream);
    ln_kernel<<<2048, 256, 0, stream>>>(T1, F(23), F(22), A2);
    ln_kernel<<<2048, 256, 0, stream>>>(A2, F(33), F(32), T1);     // decoder final norm

    int totO = 4 * 256 * 7;
    proj_kernel<<<(totO + 255) / 256, 256, 0, stream>>>(T1, F(77), F(76), (float*)d_out, 4, 512, 256);
}
